// GraphInteractionNetwork_14096082665507
// MI455X (gfx1250) — compile-verified, matrix operand order checked by reading, not run on hardware
//
#include <hip/hip_runtime.h>
#include <hip/hip_bf16.h>

typedef __attribute__((ext_vector_type(16))) _Float16 v16h;
typedef __attribute__((ext_vector_type(8)))  _Float16 v8h;
typedef __attribute__((ext_vector_type(8)))  float    v8f;

#define WMMA_F16(a, b, c) \
    __builtin_amdgcn_wmma_f32_16x16x32_f16(false, (a), false, (b), (short)0, (c), false, false)

// Forbid caching LDS/global loads across tile iterations (prevents the compiler
// from hoisting all weight fragments into VGPRs and spilling to scratch).
#define OPT_BARRIER() asm volatile("" ::: "memory")

// Convert f32 row-major weight W[K][N] into LDS fragments laid out per the CDNA5
// 16-bit B-matrix (KxN) VGPR layout: lane L holds column n = nt*16 + (L&15);
// halfs 0..7  = W[kt*32 + ((L>>4)<<3) + i][n]
// halfs 8..15 = W[kt*32 + ((L>>4)<<3) + 16 + i][n]
// Fragment (kt,nt), lane L stored contiguously at sW + ((kt*ntiles+nt)*32 + L)*16.
__device__ inline void stage_frags(const float* __restrict__ W, _Float16* sW,
                                   int N, int ktiles, int ntiles) {
    const int nent = ktiles * ntiles * 32;
    for (int idx = threadIdx.x; idx < nent; idx += blockDim.x) {
        const int f  = idx >> 5;
        const int l  = idx & 31;
        const int kt = f / ntiles;
        const int nt = f - kt * ntiles;
        const int n  = nt * 16 + (l & 15);
        const int kb = kt * 32 + ((l >> 4) << 3);
        _Float16* dst = sW + (size_t)idx * 16;
        const float* src = W + (size_t)kb * N + n;
#pragma unroll
        for (int i = 0; i < 8; ++i) {
            dst[i]     = (_Float16)src[i * N];
            dst[8 + i] = (_Float16)src[(16 + i) * N];
        }
    }
}

__device__ inline v16h make_a(float4 f0, float4 f1, float4 f2, float4 f3) {
    v16h a;
    a[0]  = (_Float16)f0.x; a[1]  = (_Float16)f0.y; a[2]  = (_Float16)f0.z; a[3]  = (_Float16)f0.w;
    a[4]  = (_Float16)f1.x; a[5]  = (_Float16)f1.y; a[6]  = (_Float16)f1.z; a[7]  = (_Float16)f1.w;
    a[8]  = (_Float16)f2.x; a[9]  = (_Float16)f2.y; a[10] = (_Float16)f2.z; a[11] = (_Float16)f2.w;
    a[12] = (_Float16)f3.x; a[13] = (_Float16)f3.y; a[14] = (_Float16)f3.z; a[15] = (_Float16)f3.w;
    return a;
}

__global__ void gin_zero_kernel(float* __restrict__ p, size_t n) {
    size_t i = (size_t)blockIdx.x * blockDim.x + threadIdx.x;
    const size_t stride = (size_t)gridDim.x * blockDim.x;
    for (; i < n; i += stride) p[i] = 0.0f;
}

// ---- Edge block: h = relu(concat[edge, nodes[recv], nodes[send]] @ We1 + be1)
//      new_edge = h @ We2 + be2 ; atomically scatter-add into agg[recv] ----
__global__ __launch_bounds__(256) void gin_edge_kernel(
    const float* __restrict__ nodes, const float* __restrict__ edges,
    const float* __restrict__ We1, const float* __restrict__ be1,
    const float* __restrict__ We2, const float* __restrict__ be2,
    const int* __restrict__ senders, const int* __restrict__ receivers,
    float* __restrict__ agg, int nEdges)
{
    __shared__ __align__(32) _Float16 sW1[48 * 512];      // 6 ktiles x 8 ntiles (48 KB)
    __shared__ __align__(32) _Float16 sW2[16 * 512];      // 4 ktiles x 4 ntiles (16 KB)
    __shared__ __align__(32) _Float16 sH[8][16 * 128];    // per-wave h tile     (32 KB)

    stage_frags(We1, sW1, 128, 6, 8);
    stage_frags(We2, sW2, 64, 4, 4);
    __syncthreads();

    const int lane   = threadIdx.x & 31;
    const int wave   = threadIdx.x >> 5;
    const int col    = lane & 15;          // N-column / A-row owned by this lane
    const int rowAdd = (lane >> 4) << 3;   // K-offset (A/B) and M-offset (C/D)

    float b1v[8], b2v[4];
#pragma unroll
    for (int nt = 0; nt < 8; ++nt) b1v[nt] = be1[nt * 16 + col];
#pragma unroll
    for (int nt = 0; nt < 4; ++nt) b2v[nt] = be2[nt * 16 + col];

    const int nTiles = nEdges >> 4;
    for (int tile = blockIdx.x * 8 + wave; tile < nTiles; tile += gridDim.x * 8) {
        OPT_BARRIER();
        const int e  = (tile << 4) + col;
        const int rv = receivers[e];
        const int sd = senders[e];
        const float* s0 = edges + (size_t)e  * 64;
        const float* s1 = nodes + (size_t)rv * 64;
        const float* s2 = nodes + (size_t)sd * 64;

        v8f acc[8];
#pragma unroll
        for (int nt = 0; nt < 8; ++nt)
#pragma unroll
            for (int i = 0; i < 8; ++i) acc[nt][i] = b1v[nt];

        // ---- layer 1: [16x192] @ [192x128], K in 6 tiles of 32 ----
        // software pipeline: global loads for kt+1 issue under kt's WMMA chain.
        // Prefetch is straight-line (kn clamped; pointer chosen via cndmask).
        float4 f0 = *(const float4*)(s0 + rowAdd);
        float4 f1 = *(const float4*)(s0 + rowAdd + 4);
        float4 f2 = *(const float4*)(s0 + rowAdd + 16);
        float4 f3 = *(const float4*)(s0 + rowAdd + 20);
#pragma unroll 1
        for (int kt = 0; kt < 6; ++kt) {
            v16h a = make_a(f0, f1, f2, f3);
            {
                const int kn = (kt < 5) ? kt + 1 : 5;     // clamp: last iter reloads (L0 hit)
                const float* s = (kn < 4) ? s1 : s2;
                s = (kn < 2) ? s0 : s;
                const int o = ((kn & 1) << 5) + rowAdd;
                f0 = *(const float4*)(s + o);
                f1 = *(const float4*)(s + o + 4);
                f2 = *(const float4*)(s + o + 16);
                f3 = *(const float4*)(s + o + 20);
            }
            const _Float16* wrow = &sW1[((kt * 8) * 32 + lane) * 16];
            // double-buffered B fragments: load runs under previous WMMA
            v16h bA = *(const v16h*)(wrow);
            v16h bB = *(const v16h*)(wrow + 512);
#pragma unroll
            for (int nt = 0; nt < 8; nt += 2) {
                acc[nt] = WMMA_F16(a, bA, acc[nt]);
                if (nt + 2 < 8) bA = *(const v16h*)(wrow + (nt + 2) * 512);
                acc[nt + 1] = WMMA_F16(a, bB, acc[nt + 1]);
                if (nt + 3 < 8) bB = *(const v16h*)(wrow + (nt + 3) * 512);
            }
        }

        // ---- relu + D-layout -> A-layout transpose through LDS (f16) ----
        _Float16* hb = sH[wave];
#pragma unroll
        for (int nt = 0; nt < 8; ++nt)
#pragma unroll
            for (int v = 0; v < 8; ++v)
                hb[(rowAdd + v) * 128 + nt * 16 + col] = (_Float16)fmaxf(acc[nt][v], 0.0f);

        // ---- layer 2: [16x128] @ [128x64] ----
        v8f acc2[4];
#pragma unroll
        for (int nt = 0; nt < 4; ++nt)
#pragma unroll
            for (int i = 0; i < 8; ++i) acc2[nt][i] = b2v[nt];

        v8h lo = *(const v8h*)&hb[col * 128 + rowAdd];
        v8h hi = *(const v8h*)&hb[col * 128 + rowAdd + 16];
#pragma unroll 1
        for (int kt = 0; kt < 4; ++kt) {
            v16h a;
#pragma unroll
            for (int i = 0; i < 8; ++i) { a[i] = lo[i]; a[8 + i] = hi[i]; }
            {
                const int kn = (kt < 3) ? kt + 1 : 3;
                const int kb = (kn << 5) + rowAdd;
                lo = *(const v8h*)&hb[col * 128 + kb];
                hi = *(const v8h*)&hb[col * 128 + kb + 16];
            }
            const _Float16* wrow = &sW2[((kt * 4) * 32 + lane) * 16];
            v16h bA = *(const v16h*)(wrow);
            v16h bB = *(const v16h*)(wrow + 512);
#pragma unroll
            for (int nt = 0; nt < 4; nt += 2) {
                acc2[nt] = WMMA_F16(a, bA, acc2[nt]);
                if (nt + 2 < 4) bA = *(const v16h*)(wrow + (nt + 2) * 512);
                acc2[nt + 1] = WMMA_F16(a, bB, acc2[nt + 1]);
                if (nt + 3 < 4) bB = *(const v16h*)(wrow + (nt + 3) * 512);
            }
        }

        // ---- scatter-add into agg (L2-resident, f32 atomics) ----
        int rcv[8];
#pragma unroll
        for (int v = 0; v < 8; ++v) rcv[v] = receivers[(tile << 4) + rowAdd + v];
#pragma unroll
        for (int nt = 0; nt < 4; ++nt)
#pragma unroll
            for (int v = 0; v < 8; ++v)
                atomicAdd(&agg[(size_t)rcv[v] * 64 + nt * 16 + col], acc2[nt][v]);
    }
}

// ---- Node block: out = relu(concat[agg, nodes] @ Wn1 + bn1) @ Wn2 + bn2 ----
__global__ __launch_bounds__(256) void gin_node_kernel(
    const float* __restrict__ nodes, const float* __restrict__ agg,
    const float* __restrict__ Wn1, const float* __restrict__ bn1,
    const float* __restrict__ Wn2, const float* __restrict__ bn2,
    float* __restrict__ out, int nNodes)
{
    __shared__ __align__(32) _Float16 sW1[32 * 512];      // 4 ktiles x 8 ntiles (32 KB)
    __shared__ __align__(32) _Float16 sW2[16 * 512];      // 4 ktiles x 4 ntiles (16 KB)
    __shared__ __align__(32) _Float16 sH[8][16 * 128];    // 32 KB

    stage_frags(Wn1, sW1, 128, 4, 8);
    stage_frags(Wn2, sW2, 64, 4, 4);
    __syncthreads();

    const int lane   = threadIdx.x & 31;
    const int wave   = threadIdx.x >> 5;
    const int col    = lane & 15;
    const int rowAdd = (lane >> 4) << 3;

    float b1v[8], b2v[4];
#pragma unroll
    for (int nt = 0; nt < 8; ++nt) b1v[nt] = bn1[nt * 16 + col];
#pragma unroll
    for (int nt = 0; nt < 4; ++nt) b2v[nt] = bn2[nt * 16 + col];

    const int nTiles = nNodes >> 4;
    for (int tile = blockIdx.x * 8 + wave; tile < nTiles; tile += gridDim.x * 8) {
        OPT_BARRIER();
        const int n = (tile << 4) + col;
        const float* s0 = agg   + (size_t)n * 64;
        const float* s1 = nodes + (size_t)n * 64;

        v8f acc[8];
#pragma unroll
        for (int nt = 0; nt < 8; ++nt)
#pragma unroll
            for (int i = 0; i < 8; ++i) acc[nt][i] = b1v[nt];

        float4 f0 = *(const float4*)(s0 + rowAdd);
        float4 f1 = *(const float4*)(s0 + rowAdd + 4);
        float4 f2 = *(const float4*)(s0 + rowAdd + 16);
        float4 f3 = *(const float4*)(s0 + rowAdd + 20);
#pragma unroll 1
        for (int kt = 0; kt < 4; ++kt) {
            v16h a = make_a(f0, f1, f2, f3);
            {
                const int kn = (kt < 3) ? kt + 1 : 3;
                const float* s = (kn < 2) ? s0 : s1;
                const int o = ((kn & 1) << 5) + rowAdd;
                f0 = *(const float4*)(s + o);
                f1 = *(const float4*)(s + o + 4);
                f2 = *(const float4*)(s + o + 16);
                f3 = *(const float4*)(s + o + 20);
            }
            const _Float16* wrow = &sW1[((kt * 8) * 32 + lane) * 16];
            v16h bA = *(const v16h*)(wrow);
            v16h bB = *(const v16h*)(wrow + 512);
#pragma unroll
            for (int nt = 0; nt < 8; nt += 2) {
                acc[nt] = WMMA_F16(a, bA, acc[nt]);
                if (nt + 2 < 8) bA = *(const v16h*)(wrow + (nt + 2) * 512);
                acc[nt + 1] = WMMA_F16(a, bB, acc[nt + 1]);
                if (nt + 3 < 8) bB = *(const v16h*)(wrow + (nt + 3) * 512);
            }
        }

        _Float16* hb = sH[wave];
#pragma unroll
        for (int nt = 0; nt < 8; ++nt)
#pragma unroll
            for (int v = 0; v < 8; ++v)
                hb[(rowAdd + v) * 128 + nt * 16 + col] = (_Float16)fmaxf(acc[nt][v], 0.0f);

        v8f acc2[4];
#pragma unroll
        for (int nt = 0; nt < 4; ++nt)
#pragma unroll
            for (int i = 0; i < 8; ++i) acc2[nt][i] = b2v[nt];

        v8h lo = *(const v8h*)&hb[col * 128 + rowAdd];
        v8h hi = *(const v8h*)&hb[col * 128 + rowAdd + 16];
#pragma unroll 1
        for (int kt = 0; kt < 4; ++kt) {
            v16h a;
#pragma unroll
            for (int i = 0; i < 8; ++i) { a[i] = lo[i]; a[8 + i] = hi[i]; }
            {
                const int kn = (kt < 3) ? kt + 1 : 3;
                const int kb = (kn << 5) + rowAdd;
                lo = *(const v8h*)&hb[col * 128 + kb];
                hi = *(const v8h*)&hb[col * 128 + kb + 16];
            }
            const _Float16* wrow = &sW2[((kt * 4) * 32 + lane) * 16];
            v16h bA = *(const v16h*)(wrow);
            v16h bB = *(const v16h*)(wrow + 512);
#pragma unroll
            for (int nt = 0; nt < 4; nt += 2) {
                acc2[nt] = WMMA_F16(a, bA, acc2[nt]);
                if (nt + 2 < 4) bA = *(const v16h*)(wrow + (nt + 2) * 512);
                acc2[nt + 1] = WMMA_F16(a, bB, acc2[nt + 1]);
                if (nt + 3 < 4) bB = *(const v16h*)(wrow + (nt + 3) * 512);
            }
        }

#pragma unroll
        for (int nt = 0; nt < 4; ++nt)
#pragma unroll
            for (int v = 0; v < 8; ++v)
                out[(size_t)((tile << 4) + rowAdd + v) * 64 + nt * 16 + col] = acc2[nt][v];
    }
}

extern "C" void kernel_launch(void* const* d_in, const int* in_sizes, int n_in,
                              void* d_out, int out_size, void* d_ws, size_t ws_size,
                              hipStream_t stream) {
    const float* nodes     = (const float*)d_in[0];
    const float* edges     = (const float*)d_in[1];
    const float* We1       = (const float*)d_in[2];
    const float* be1       = (const float*)d_in[3];
    const float* We2       = (const float*)d_in[4];
    const float* be2       = (const float*)d_in[5];
    const float* Wn1       = (const float*)d_in[6];
    const float* bn1       = (const float*)d_in[7];
    const float* Wn2       = (const float*)d_in[8];
    const float* bn2       = (const float*)d_in[9];
    const int*   senders   = (const int*)d_in[10];
    const int*   receivers = (const int*)d_in[11];

    const int nNodes = in_sizes[0] / 64;
    const int nEdges = in_sizes[1] / 64;

    float* agg = (float*)d_ws;        // [nNodes, 64] accumulator
    float* out = (float*)d_out;       // [nNodes, 64]

    gin_zero_kernel<<<512, 256, 0, stream>>>(agg, (size_t)nNodes * 64);
    gin_edge_kernel<<<800, 256, 0, stream>>>(nodes, edges, We1, be1, We2, be2,
                                             senders, receivers, agg, nEdges);
    gin_node_kernel<<<256, 256, 0, stream>>>(nodes, agg, Wn1, bn1, Wn2, bn2,
                                             out, nNodes);
}
